// GNNLayer_52226802319684
// MI455X (gfx1250) — matrix operand order repaired and checked by reading, not compiled, not run on hardware
//
#include <hip/hip_runtime.h>

// ---------------------------------------------------------------------------
// GNN layer for MI455X (gfx1250): bf16 WMMA GEMMs + fused LN/ReLU + L2 atomics
// Shapes (fixed by setup_inputs): B=2, N=10000, H=256, 2H=512, E=320000
//
// Edge kernel: 64-edge M-blocking -> each wave runs 4 wmma per B-fragment,
// cutting weight re-reads from L2 by 4x and loads-per-wmma from 4 to 2.5.
// ---------------------------------------------------------------------------

typedef __attribute__((ext_vector_type(16))) __bf16 v16bf;
typedef __attribute__((ext_vector_type(8)))  float  v8f;

union Frag16 {
    uint4 q[2];   // 32 bytes
    v16bf v;      // 16 x bf16
};

__device__ __forceinline__ unsigned int bf16_rne(float f) {
    unsigned int u = __float_as_uint(f);
    u += 0x7FFFu + ((u >> 16) & 1u);
    return u >> 16;
}
__device__ __forceinline__ unsigned int pack_bf16x2(float a, float b) {
    return bf16_rne(a) | (bf16_rne(b) << 16);
}

// ---------------------------------------------------------------------------
// prep kernels
// ---------------------------------------------------------------------------
__global__ void cvt_f32_to_bf16(const float* __restrict__ src,
                                unsigned short* __restrict__ dst, int n) {
    int i = blockIdx.x * blockDim.x + threadIdx.x;
    if (i < n) dst[i] = (unsigned short)bf16_rne(src[i]);
}

__global__ void zero_f32(float* __restrict__ p, int n) {
    int i = blockIdx.x * blockDim.x + threadIdx.x;
    if (i < n) p[i] = 0.0f;
}

// ---------------------------------------------------------------------------
// Edge kernel: workgroup = 64 edges x 256 outputs, 16 waves (512 threads).
//   A = concat(h[src], h[dst])  (64 x 512 bf16, staged in LDS, 64 KB)
//   B = W_e bf16 [512][256] row-major (== CDNA5 B-fragment layout)
//   Each wave: 4 M-tiles x 1 N-tile -> 4 wmma share one B fragment.
//   LDS is reused (aliased) for the 64x256 f32 message tile after a barrier.
// ---------------------------------------------------------------------------
__global__ __launch_bounds__(512)
void edge_kernel(const float* __restrict__ h,              // [B][N][256]
                 const long long* __restrict__ eidx,       // [2][E]
                 const unsigned short* __restrict__ Wbf,   // [512][256] bf16
                 const float* __restrict__ bias,           // [256]
                 const float* __restrict__ gamma,
                 const float* __restrict__ beta,
                 float* __restrict__ agg,                  // [B][N][256]
                 int N, int E) {
    constexpr int H = 256;
    __shared__ __align__(16) unsigned char smem[64 * 1024];
    unsigned short* sA   = (unsigned short*)smem;   // [64][512] bf16 (phase 1)
    float*          sMsg = (float*)smem;            // [64][256] f32  (phase 2)

    const int b    = blockIdx.y;
    const int e0   = blockIdx.x * 64;
    const int t    = threadIdx.x;
    const int lane = t & 31;
    const int wave = t >> 5;

    // ---- stage A tile: each thread gathers 4 chunks of 16 f32 -> bf16 LDS
#pragma unroll
    for (int j = 0; j < 4; ++j) {
        const int r = (t >> 5) + j * 16;   // edge row 0..63
        const int c = t & 31;              // chunk 0..31 (16 elems each)
        const long long ge   = (long long)(e0 + r);
        const long long node = (c < 16) ? eidx[ge] : eidx[(long long)E + ge];
        const int cc = (c < 16) ? c : (c - 16);
        const float4* s4 =
            (const float4*)(h + ((size_t)b * N + (size_t)node) * H + cc * 16);
        float4 f0 = s4[0], f1 = s4[1], f2 = s4[2], f3 = s4[3];
        uint4 o0, o1;
        o0.x = pack_bf16x2(f0.x, f0.y); o0.y = pack_bf16x2(f0.z, f0.w);
        o0.z = pack_bf16x2(f1.x, f1.y); o0.w = pack_bf16x2(f1.z, f1.w);
        o1.x = pack_bf16x2(f2.x, f2.y); o1.y = pack_bf16x2(f2.z, f2.w);
        o1.z = pack_bf16x2(f3.x, f3.y); o1.w = pack_bf16x2(f3.z, f3.w);
        uint4* dstp = (uint4*)(sA + r * 512 + c * 16);
        dstp[0] = o0; dstp[1] = o1;
    }
    __syncthreads();

    // ---- GEMM: wave computes rows {0..63} x cols [n0, n0+16), K = 512
    const int n0   = wave * 16;
    const int row  = lane & 15;
    const int half = lane >> 4;

    v8f acc0 = {}, acc1 = {}, acc2 = {}, acc3 = {};
#pragma unroll 4
    for (int kk = 0; kk < 16; ++kk) {
        const int k0 = kk * 32;
        // one B fragment, shared by 4 wmma (lane <-> N, elements <-> K in table order)
        Frag16 bm;
        const unsigned short* pb = Wbf + (size_t)(k0 + half * 8) * H + n0 + row;
#pragma unroll
        for (int i = 0; i < 4; ++i) {
            ((unsigned int*)&bm.q[0])[i] = pb[(2 * i) * H] | ((unsigned int)pb[(2 * i + 1) * H] << 16);
            ((unsigned int*)&bm.q[1])[i] = pb[(16 + 2 * i) * H] | ((unsigned int)pb[(17 + 2 * i) * H] << 16);
        }
        // four A fragments (M-tiles at rows 0/16/32/48)
        const char* aB = (const char*)sA + half * 16 + k0 * 2;
        Frag16 a0, a1, a2, a3;
        a0.q[0] = *(const uint4*)(aB + (row     ) * 1024);
        a0.q[1] = *(const uint4*)(aB + (row     ) * 1024 + 32);
        a1.q[0] = *(const uint4*)(aB + (row + 16) * 1024);
        a1.q[1] = *(const uint4*)(aB + (row + 16) * 1024 + 32);
        a2.q[0] = *(const uint4*)(aB + (row + 32) * 1024);
        a2.q[1] = *(const uint4*)(aB + (row + 32) * 1024 + 32);
        a3.q[0] = *(const uint4*)(aB + (row + 48) * 1024);
        a3.q[1] = *(const uint4*)(aB + (row + 48) * 1024 + 32);
        acc0 = __builtin_amdgcn_wmma_f32_16x16x32_bf16(
            false, a0.v, false, bm.v, (short)0, acc0, false, false);
        acc1 = __builtin_amdgcn_wmma_f32_16x16x32_bf16(
            false, a1.v, false, bm.v, (short)0, acc1, false, false);
        acc2 = __builtin_amdgcn_wmma_f32_16x16x32_bf16(
            false, a2.v, false, bm.v, (short)0, acc2, false, false);
        acc3 = __builtin_amdgcn_wmma_f32_16x16x32_bf16(
            false, a3.v, false, bm.v, (short)0, acc3, false, false);
    }
    __syncthreads();   // all waves done reading sA; safe to alias as sMsg

    // ---- D + bias -> sMsg (C/D layout: lane -> col, vgpr r -> row)
    {
        const int col = lane & 15;
        const int mb  = half * 8;
        const float bv = bias[n0 + col];
#pragma unroll
        for (int r = 0; r < 8; ++r) {
            sMsg[(mb + r     ) * 256 + n0 + col] = acc0[r] + bv;
            sMsg[(mb + r + 16) * 256 + n0 + col] = acc1[r] + bv;
            sMsg[(mb + r + 32) * 256 + n0 + col] = acc2[r] + bv;
            sMsg[(mb + r + 48) * 256 + n0 + col] = acc3[r] + bv;
        }
    }
    __syncthreads();

    // ---- LayerNorm + ReLU + atomic scatter-add (4 edge rows per wave)
#pragma unroll
    for (int j = 0; j < 4; ++j) {
        const int m = wave * 4 + j;    // edge row 0..63
        const long long ge = (long long)(e0 + m);
        const long long dn = eidx[(long long)E + ge];
        float s = 0.f, s2 = 0.f, x[8];
#pragma unroll
        for (int i = 0; i < 8; ++i) {
            x[i] = sMsg[m * 256 + lane + i * 32];
            s += x[i]; s2 += x[i] * x[i];
        }
#pragma unroll
        for (int off = 16; off > 0; off >>= 1) {
            s  += __shfl_xor(s,  off, 32);
            s2 += __shfl_xor(s2, off, 32);
        }
        const float mean = s * (1.0f / 256.0f);
        const float var  = s2 * (1.0f / 256.0f) - mean * mean;
        const float rstd = rsqrtf(var + 1e-5f);
        float* aggRow = agg + ((size_t)b * N + (size_t)dn) * H;
#pragma unroll
        for (int i = 0; i < 8; ++i) {
            const int c = lane + i * 32;
            float y = (x[i] - mean) * rstd * gamma[c] + beta[c];
            y = fmaxf(y, 0.0f);
            atomicAdd(aggRow + c, y);   // global_atomic_add_f32, L2-resident
        }
    }
}

// ---------------------------------------------------------------------------
// Node kernel: per workgroup = 16 nodes x 256 outputs (N=10000 not /64).
//   A = concat(h, aggregated); out = ReLU(LN(A @ W_n + b)) + h
// ---------------------------------------------------------------------------
__global__ __launch_bounds__(512)
void node_kernel(const float* __restrict__ h,
                 const float* __restrict__ agg,
                 const unsigned short* __restrict__ Wbf,   // [512][256] bf16
                 const float* __restrict__ bias,
                 const float* __restrict__ gamma,
                 const float* __restrict__ beta,
                 float* __restrict__ out, int N) {
    constexpr int H = 256;
    __shared__ unsigned short sA[16 * 512];
    __shared__ float sMsg[16 * 256];

    const int b    = blockIdx.y;
    const int v0   = blockIdx.x * 16;     // first node of tile
    const int t    = threadIdx.x;
    const int lane = t & 31;
    const int wave = t >> 5;

    // ---- stage A tile
    {
        const int r = t >> 5;
        const int c = t & 31;
        const int node = v0 + r;
        const int cc = (c < 16) ? c : (c - 16);
        const float* srcBase = (c < 16) ? h : agg;
        const float4* s4 =
            (const float4*)(srcBase + ((size_t)b * N + node) * H + cc * 16);
        float4 f0 = s4[0], f1 = s4[1], f2 = s4[2], f3 = s4[3];
        uint4 o0, o1;
        o0.x = pack_bf16x2(f0.x, f0.y); o0.y = pack_bf16x2(f0.z, f0.w);
        o0.z = pack_bf16x2(f1.x, f1.y); o0.w = pack_bf16x2(f1.z, f1.w);
        o1.x = pack_bf16x2(f2.x, f2.y); o1.y = pack_bf16x2(f2.z, f2.w);
        o1.z = pack_bf16x2(f3.x, f3.y); o1.w = pack_bf16x2(f3.z, f3.w);
        uint4* dstp = (uint4*)(sA + r * 512 + c * 16);
        dstp[0] = o0; dstp[1] = o1;
    }
    __syncthreads();

    const int n0   = wave * 16;
    const int row  = lane & 15;
    const int half = lane >> 4;
    const char* aBase = (const char*)sA + row * 1024 + half * 16;

    v8f acc = {};
#pragma unroll
    for (int kk = 0; kk < 16; ++kk) {
        const int k0 = kk * 32;
        Frag16 a, bm;
        a.q[0] = *(const uint4*)(aBase + k0 * 2);
        a.q[1] = *(const uint4*)(aBase + k0 * 2 + 32);
        const unsigned short* pb = Wbf + (size_t)(k0 + half * 8) * H + n0 + row;
#pragma unroll
        for (int i = 0; i < 4; ++i) {
            ((unsigned int*)&bm.q[0])[i] = pb[(2 * i) * H] | ((unsigned int)pb[(2 * i + 1) * H] << 16);
            ((unsigned int*)&bm.q[1])[i] = pb[(16 + 2 * i) * H] | ((unsigned int)pb[(17 + 2 * i) * H] << 16);
        }
        acc = __builtin_amdgcn_wmma_f32_16x16x32_bf16(
            false, a.v, false, bm.v, (short)0, acc, false, false);
    }

    {
        const int col = lane & 15;
        const int mb  = half * 8;
        const float bv = bias[n0 + col];
#pragma unroll
        for (int r = 0; r < 8; ++r)
            sMsg[(mb + r) * 256 + n0 + col] = acc[r] + bv;
    }
    __syncthreads();

    // ---- LayerNorm + ReLU + residual, one wave per node row
    {
        const int m    = wave;
        const int node = v0 + m;
        float s = 0.f, s2 = 0.f, x[8];
#pragma unroll
        for (int i = 0; i < 8; ++i) {
            x[i] = sMsg[m * 256 + lane + i * 32];
            s += x[i]; s2 += x[i] * x[i];
        }
#pragma unroll
        for (int off = 16; off > 0; off >>= 1) {
            s  += __shfl_xor(s,  off, 32);
            s2 += __shfl_xor(s2, off, 32);
        }
        const float mean = s * (1.0f / 256.0f);
        const float var  = s2 * (1.0f / 256.0f) - mean * mean;
        const float rstd = rsqrtf(var + 1e-5f);
        const float* hRow = h   + ((size_t)b * N + node) * H;
        float*       oRow = out + ((size_t)b * N + node) * H;
#pragma unroll
        for (int i = 0; i < 8; ++i) {
            const int c = lane + i * 32;
            float y = (x[i] - mean) * rstd * gamma[c] + beta[c];
            y = fmaxf(y, 0.0f);
            oRow[c] = y + hRow[c];
        }
    }
}

// ---------------------------------------------------------------------------
// host launcher
// ---------------------------------------------------------------------------
extern "C" void kernel_launch(void* const* d_in, const int* in_sizes, int n_in,
                              void* d_out, int out_size, void* d_ws, size_t ws_size,
                              hipStream_t stream) {
    const float*     h    = (const float*)d_in[0];
    const long long* eidx = (const long long*)d_in[1];   // int64 edge_index
    const float*     W_e  = (const float*)d_in[2];
    const float*     b_e  = (const float*)d_in[3];
    const float*     g_e  = (const float*)d_in[4];
    const float*     be_e = (const float*)d_in[5];
    const float*     W_n  = (const float*)d_in[6];
    const float*     b_n  = (const float*)d_in[7];
    const float*     g_n  = (const float*)d_in[8];
    const float*     be_n = (const float*)d_in[9];
    float* out = (float*)d_out;

    const int H = 256;
    const int B = 2;
    const int E = in_sizes[1] / 2;      // 320000 (divisible by 64)
    const int N = in_sizes[0] / (B * H);// 10000  (divisible by 16)

    // workspace layout: W_e bf16 | W_n bf16 | aggregated f32
    unsigned short* We_bf = (unsigned short*)d_ws;        // 2H*H ushorts
    unsigned short* Wn_bf = We_bf + 2 * H * H;
    float* agg = (float*)(Wn_bf + 2 * H * H);             // B*N*H floats

    const int wElems = 2 * H * H;                         // 131072
    cvt_f32_to_bf16<<<(wElems + 255) / 256, 256, 0, stream>>>(W_e, We_bf, wElems);
    cvt_f32_to_bf16<<<(wElems + 255) / 256, 256, 0, stream>>>(W_n, Wn_bf, wElems);

    const int aggElems = B * N * H;
    zero_f32<<<(aggElems + 255) / 256, 256, 0, stream>>>(agg, aggElems);

    dim3 eg(E / 64, B);
    edge_kernel<<<eg, 512, 0, stream>>>(h, eidx, We_bf, b_e, g_e, be_e, agg, N, E);

    dim3 ng(N / 16, B);
    node_kernel<<<ng, 512, 0, stream>>>(h, agg, Wn_bf, b_n, g_n, be_n, out, N);
}
